// CausalSlidingWindowAttention_11647951306845
// MI455X (gfx1250) — matrix operand order repaired and checked by reading, not compiled, not run on hardware
//
#include <hip/hip_runtime.h>

typedef __attribute__((ext_vector_type(16))) __bf16 v16bf;
typedef __attribute__((ext_vector_type(8)))  float  v8f;

#define B_SZ   2
#define T_SZ   2048
#define NH     16
#define HD     64
#define D_SZ   (NH * HD)        // 1024

__device__ __forceinline__ v8f wmma_bf16(v16bf a, v16bf b, v8f c) {
    return __builtin_amdgcn_wmma_f32_16x16x32_bf16(false, a, false, b, (short)0, c, false, false);
}

// A-fragment (16x32, 16-bit) per ISA 7.12.2: lanes 0-15 hold K 0-7 (e0..7) and K16-23 (e8..15);
// lanes 16-31 hold K 8-15 and K 24-31.
__device__ __forceinline__ void a_map(int k, int& h, int& e) {
    h = (k >> 3) & 1;
    e = (k & 7) | ((k >> 4) << 3);
}
// Contiguous k order this code uses for A and B fragments alike: lanes 0-15 hold K 0-15 (e=k), lanes 16-31 hold K 16-31.
__device__ __forceinline__ void b_map(int k, int& h, int& e) {
    h = k >> 4;
    e = k & 15;
}

__device__ __forceinline__ unsigned lds_byte_addr(const void* p) {
    // Flat LDS pointer: addr[31:0] is the wave-relative LDS offset (ISA 10.2 aperture calc).
    return (unsigned)(size_t)p;
}

// ---------------------------------------------------------------------------
// Generic GEMM: C[M,N] = A[M,K] @ W[K,N] + bias, fp32 accumulate via bf16 WMMA.
// Block tile 64x128, 8 waves arranged 2(M) x 4(N), each wave -> 32x32 output.
// ---------------------------------------------------------------------------
template<bool A_BF16>
__global__ __launch_bounds__(256) void gemm_kernel(const void* __restrict__ A_,
                                                   const float* __restrict__ W,
                                                   const float* __restrict__ bias,
                                                   float* __restrict__ C,
                                                   int M, int N, int K) {
    __shared__ __align__(32) __bf16 lds_a[4 * 32 * 16];  // 4 M-subtiles, fragment-packed
    __shared__ __align__(32) __bf16 lds_b[8 * 32 * 16];  // 8 N-subtiles, fragment-packed

    const int tid   = threadIdx.x;
    const int lane  = tid & 31;
    const int wave  = tid >> 5;
    const int lr    = lane & 15;
    const int halfid = lane >> 4;
    const int waveM = wave >> 2;   // 0..1
    const int waveN = wave & 3;    // 0..3
    const int blockM = blockIdx.y * 64;
    const int blockN = blockIdx.x * 128;

    v8f acc[2][2] = {};

    const int ar     = tid >> 2;          // 0..63  (A staging row)
    const int akbase = (tid & 3) * 8;     // 0,8,16,24
    const int bkk    = tid >> 3;          // 0..31  (B staging k)
    const int bnbase = (tid & 7) * 16;    // 0..112

    for (int k0 = 0; k0 < K; k0 += 32) {
        // --- stage A tile (64 x 32) ---
        #pragma unroll
        for (int j = 0; j < 8; ++j) {
            const int k = akbase + j;
            float v;
            if (A_BF16) v = (float)((const __bf16*)A_)[(size_t)(blockM + ar) * K + k0 + k];
            else        v = ((const float*)A_)[(size_t)(blockM + ar) * K + k0 + k];
            int h, e; b_map(k, h, e);
            lds_a[(((ar >> 4) * 32) + ((ar & 15) + 16 * h)) * 16 + e] = (__bf16)v;
        }
        // --- stage B tile (32 x 128) ---
        {
            int h, e; b_map(bkk, h, e);
            #pragma unroll
            for (int j = 0; j < 16; ++j) {
                const int n = bnbase + j;
                float v = W[(size_t)(k0 + bkk) * N + blockN + n];
                lds_b[(((n >> 4) * 32) + ((n & 15) + 16 * h)) * 16 + e] = (__bf16)v;
            }
        }
        // --- prefetch next K-step tiles into cache while this one computes ---
        if (k0 + 32 < K) {
            if (A_BF16)
                __builtin_prefetch(&((const __bf16*)A_)[(size_t)(blockM + ar) * K + k0 + 32 + akbase], 0, 0);
            else
                __builtin_prefetch(&((const float*)A_)[(size_t)(blockM + ar) * K + k0 + 32 + akbase], 0, 0);
            __builtin_prefetch(&W[(size_t)(k0 + 32 + bkk) * N + blockN + bnbase], 0, 0);
        }
        __syncthreads();

        v16bf afrag[2], bfrag[2];
        #pragma unroll
        for (int i = 0; i < 2; ++i)
            afrag[i] = *(const v16bf*)&lds_a[(((waveM * 2 + i) * 32) + lane) * 16];
        #pragma unroll
        for (int j = 0; j < 2; ++j)
            bfrag[j] = *(const v16bf*)&lds_b[(((waveN * 2 + j) * 32) + lane) * 16];

        #pragma unroll
        for (int i = 0; i < 2; ++i)
            #pragma unroll
            for (int j = 0; j < 2; ++j)
                acc[i][j] = wmma_bf16(afrag[i], bfrag[j], acc[i][j]);

        __syncthreads();
    }

    // --- epilogue: bias + store ---
    #pragma unroll
    for (int i = 0; i < 2; ++i) {
        #pragma unroll
        for (int j = 0; j < 2; ++j) {
            const int col = blockN + waveN * 32 + j * 16 + lr;
            const float bv = bias[col];
            #pragma unroll
            for (int er = 0; er < 8; ++er) {
                const int row = blockM + waveM * 32 + i * 16 + er + 8 * halfid;
                C[(size_t)row * N + col] = acc[i][j][er] + bv;
            }
        }
    }
}

// ---------------------------------------------------------------------------
// RoPE + split: qkv[B*T, 3*D] fp32 -> Q/K/V bf16 [B*H][T][64] + cache fp32 out.
// One thread per (b,t,h,pair).
// ---------------------------------------------------------------------------
__global__ __launch_bounds__(256) void rope_kernel(const float* __restrict__ qkv,
                                                   const int* __restrict__ offset_p,
                                                   __bf16* __restrict__ Qo,
                                                   __bf16* __restrict__ Ko,
                                                   __bf16* __restrict__ Vo,
                                                   float* __restrict__ cache) {
    const int idx = blockIdx.x * blockDim.x + threadIdx.x;   // 0 .. 2^21-1
    const int d2 = idx & 31;
    const int h  = (idx >> 5) & 15;
    const int t  = (idx >> 9) & 2047;
    const int b  = idx >> 20;

    const int off = offset_p[0];
    // inv_freq = 10000^(-2*d2/64)
    const float inv = __expf(-((float)(2 * d2) / 64.0f) * 9.2103403719761836f);
    const float ang = (float)(t + off) * inv;
    const float c = __cosf(ang);
    const float s = __sinf(ang);

    const size_t rowbase = (size_t)(b * T_SZ + t) * (3 * D_SZ);
    const int col = h * HD + 2 * d2;
    const float q1 = qkv[rowbase + col],            q2 = qkv[rowbase + col + 1];
    const float k1 = qkv[rowbase + D_SZ + col],     k2 = qkv[rowbase + D_SZ + col + 1];
    const float v1 = qkv[rowbase + 2 * D_SZ + col], v2 = qkv[rowbase + 2 * D_SZ + col + 1];

    const float qo1 = q1 * c - q2 * s, qo2 = q1 * s + q2 * c;
    const float ko1 = k1 * c - k2 * s, ko2 = k1 * s + k2 * c;

    const size_t qi = (((size_t)(b * NH + h)) * T_SZ + t) * HD + 2 * d2;
    Qo[qi] = (__bf16)qo1;  Qo[qi + 1] = (__bf16)qo2;
    Ko[qi] = (__bf16)ko1;  Ko[qi + 1] = (__bf16)ko2;
    Vo[qi] = (__bf16)v1;   Vo[qi + 1] = (__bf16)v2;

    // new_cache[b][0][t][h][:]=k_roped, [b][1][t][h][:]=v
    const size_t ck = ((((size_t)b * 2 + 0) * T_SZ + t) * NH + h) * HD + 2 * d2;
    const size_t cv = ((((size_t)b * 2 + 1) * T_SZ + t) * NH + h) * HD + 2 * d2;
    cache[ck] = ko1; cache[ck + 1] = ko2;
    cache[cv] = v1;  cache[cv + 1] = v2;
}

// ---------------------------------------------------------------------------
// Flash-style causal attention. grid = (T/128, B*H), 256 threads = 8 waves.
// Wave w owns query rows qBase + w*16 .. +15. Key step = 32.
// K tile is staged RAW row-major via global_load_async_to_lds_b128: the S-GEMM
// B-fragment for lane (lr,halfid), subtile ns, kstep s is exactly the 32
// contiguous bytes K_tile[ns*16+lr][s*32 + halfid*16 .. +15].
// ---------------------------------------------------------------------------
__global__ __launch_bounds__(256) void attn_kernel(const __bf16* __restrict__ Q,
                                                   const __bf16* __restrict__ Kmat,
                                                   const __bf16* __restrict__ Vmat,
                                                   __bf16* __restrict__ ctx) {
    const int bh    = blockIdx.y;          // b*16 + h
    const int b     = bh >> 4;
    const int head  = bh & 15;
    const int qBase = blockIdx.x * 128;
    const int tid   = threadIdx.x;
    const int lane  = tid & 31;
    const int wave  = tid >> 5;
    const int lr    = lane & 15;
    const int halfid = lane >> 4;
    const float scale = 0.125f;            // 1/sqrt(64)

    __shared__ __align__(32) __bf16 lds_kraw[32 * 64];     // raw row-major K tile (4KB)
    __shared__ __align__(32) __bf16 lds_v[4][32 * 16];     // [hd-nsub], fragment-packed
    __shared__ __align__(32) __bf16 lds_p[8][32 * 16];     // per-wave P transpose

    const __bf16* Qb = Q    + (size_t)bh * T_SZ * HD;
    const __bf16* Kb = Kmat + (size_t)bh * T_SZ * HD;
    const __bf16* Vb = Vmat + (size_t)bh * T_SZ * HD;

    // Preload this wave's Q A-fragments (rows fixed for whole kernel).
    const int qrow = qBase + wave * 16 + lr;
    v16bf qfrag[2];
    #pragma unroll
    for (int s = 0; s < 2; ++s) {
        const __bf16* p = Qb + (size_t)qrow * HD + s * 32 + halfid * 16;
        ((uint4*)&qfrag[s])[0] = *(const uint4*)p;
        ((uint4*)&qfrag[s])[1] = *(const uint4*)(p + 8);
    }

    float m_row[8], l_row[8];
    #pragma unroll
    for (int e = 0; e < 8; ++e) { m_row[e] = -1e30f; l_row[e] = 0.0f; }
    v8f o[4] = {};

    const unsigned kraw_base = lds_byte_addr(&lds_kraw[0]);

    const int nsteps = (qBase + 128) / 32;   // uniform across block -> barriers safe
    for (int st = 0; st < nsteps; ++st) {
        const int keyBase = st * 32;

        // ---- async raw stage of K tile (32x64 bf16 = 4KB), 16B per lane ----
        {
            const unsigned byteoff = (unsigned)tid * 16u;          // 0..4080
            const unsigned loff    = kraw_base + byteoff;
            const unsigned goff    = (unsigned)keyBase * 128u + byteoff;
            asm volatile("global_load_async_to_lds_b128 %0, %1, %2"
                         :: "v"(loff), "v"(goff), "s"(Kb) : "memory");
        }
        // ---- cooperative stage of V tile (32x64) into B-fragment layout ----
        {
            const int idx = tid * 8;           // 0..2047
            const int kk  = idx >> 6;          // key row 0..31
            const int d0  = idx & 63;          // hd base (multiple of 8)
            __bf16 tmp[8];
            *(uint4*)tmp = *(const uint4*)(Vb + (size_t)(keyBase + kk) * HD + d0);
            int hv, ev; b_map(kk, hv, ev);
            #pragma unroll
            for (int j = 0; j < 8; ++j) {
                const int d = d0 + j;
                lds_v[d >> 4][((d & 15) + 16 * hv) * 16 + ev] = tmp[j];
            }
        }
        asm volatile("s_wait_asynccnt 0x0" ::: "memory");
        __syncthreads();

        // ---- S = Q @ K^T (16 x 32), B-fragments read raw from lds_kraw ----
        v8f S[2] = {};
        #pragma unroll
        for (int s = 0; s < 2; ++s) {
            v16bf kf0 = *(const v16bf*)&lds_kraw[(lr)      * 64 + s * 32 + halfid * 16];
            v16bf kf1 = *(const v16bf*)&lds_kraw[(16 + lr) * 64 + s * 32 + halfid * 16];
            S[0] = wmma_bf16(qfrag[s], kf0, S[0]);
            S[1] = wmma_bf16(qfrag[s], kf1, S[1]);
        }

        // ---- online softmax (rows live in e + 8*halfid, cols across 16 lanes) ----
        float p0v[8], p1v[8], scl[8];
        #pragma unroll
        for (int e = 0; e < 8; ++e) {
            const int row = qBase + wave * 16 + e + 8 * halfid;
            float s0 = S[0][e] * scale;
            float s1 = S[1][e] * scale;
            if (keyBase + lr > row)      s0 = -1e30f;
            if (keyBase + 16 + lr > row) s1 = -1e30f;
            float mx = fmaxf(s0, s1);
            #pragma unroll
            for (int m = 1; m < 16; m <<= 1) mx = fmaxf(mx, __shfl_xor(mx, m, 32));
            const float mn = fmaxf(m_row[e], mx);
            const float p0 = __expf(s0 - mn);
            const float p1 = __expf(s1 - mn);
            float sum = p0 + p1;
            #pragma unroll
            for (int m = 1; m < 16; m <<= 1) sum += __shfl_xor(sum, m, 32);
            const float sc = __expf(m_row[e] - mn);
            l_row[e] = l_row[e] * sc + sum;
            m_row[e] = mn;
            p0v[e] = p0; p1v[e] = p1; scl[e] = sc;
        }
        #pragma unroll
        for (int j = 0; j < 4; ++j)
            #pragma unroll
            for (int e = 0; e < 8; ++e) o[j][e] *= scl[e];

        // ---- transpose P (C layout -> A fragment) through per-wave LDS ----
        #pragma unroll
        for (int ns = 0; ns < 2; ++ns) {
            #pragma unroll
            for (int e = 0; e < 8; ++e) {
                const int mrow = e + 8 * halfid;       // 0..15
                const int k    = ns * 16 + lr;         // key col 0..31
                int h, ee; b_map(k, h, ee);
                lds_p[wave][(mrow + 16 * h) * 16 + ee] = (__bf16)(ns ? p1v[e] : p0v[e]);
            }
        }
        asm volatile("s_wait_dscnt 0x0" ::: "memory");   // within-wave LDS store->load order
        v16bf pf = *(const v16bf*)&lds_p[wave][lane * 16];

        // ---- O += P @ V ----
        #pragma unroll
        for (int j = 0; j < 4; ++j) {
            v16bf vf = *(const v16bf*)&lds_v[j][lane * 16];
            o[j] = wmma_bf16(pf, vf, o[j]);
        }
        __syncthreads();
    }

    // ---- normalize and store ctx (bf16, [B*T][D]) ----
    float rcp[8];
    #pragma unroll
    for (int e = 0; e < 8; ++e) rcp[e] = 1.0f / l_row[e];
    #pragma unroll
    for (int j = 0; j < 4; ++j) {
        #pragma unroll
        for (int e = 0; e < 8; ++e) {
            const int row = qBase + wave * 16 + e + 8 * halfid;
            const int col = head * HD + j * 16 + lr;
            ctx[(size_t)(b * T_SZ + row) * D_SZ + col] = (__bf16)(o[j][e] * rcp[e]);
        }
    }
}

// ---------------------------------------------------------------------------
extern "C" void kernel_launch(void* const* d_in, const int* in_sizes, int n_in,
                              void* d_out, int out_size, void* d_ws, size_t ws_size,
                              hipStream_t stream) {
    const float* x    = (const float*)d_in[0];   // [2,2048,1024]
    const float* Wqkv = (const float*)d_in[1];   // [1024,3072]
    const float* bqkv = (const float*)d_in[2];   // [3072]
    const float* Wout = (const float*)d_in[3];   // [1024,1024]
    const float* bout = (const float*)d_in[4];   // [1024]
    const int*   off  = (const int*)d_in[5];     // scalar

    float* out   = (float*)d_out;                       // [2,2048,1024]
    float* cache = out + (size_t)B_SZ * T_SZ * D_SZ;    // [2,2,2048,16,64]

    char* ws = (char*)d_ws;
    const size_t M = (size_t)B_SZ * T_SZ;               // 4096
    float*  qkv = (float*)ws;                           // M x 3072 fp32 (48 MiB)
    __bf16* Qb  = (__bf16*)(ws + M * 3 * D_SZ * sizeof(float));
    __bf16* Kb  = Qb + M * D_SZ;
    __bf16* Vb  = Kb + M * D_SZ;
    __bf16* ctx = Vb + M * D_SZ;                        // M x 1024 bf16

    // 1) qkv = x @ Wqkv + bqkv
    gemm_kernel<false><<<dim3(3 * D_SZ / 128, M / 64), 256, 0, stream>>>(
        x, Wqkv, bqkv, qkv, (int)M, 3 * D_SZ, D_SZ);

    // 2) RoPE + split + cache output
    rope_kernel<<<(B_SZ * T_SZ * NH * 32) / 256, 256, 0, stream>>>(
        qkv, off, Qb, Kb, Vb, cache);

    // 3) causal attention -> ctx
    attn_kernel<<<dim3(T_SZ / 128, B_SZ * NH), 256, 0, stream>>>(Qb, Kb, Vb, ctx);

    // 4) out = ctx @ Wout + bout
    gemm_kernel<true><<<dim3(D_SZ / 128, M / 64), 256, 0, stream>>>(
        ctx, Wout, bout, out, (int)M, D_SZ, D_SZ);
}
